// WorkbenchCausalSelfAttention_66211215835396
// MI455X (gfx1250) — matrix operand order repaired and checked by reading, not compiled, not run on hardware
//
#include <hip/hip_runtime.h>
#include <stdint.h>

// ---------------------------------------------------------------------------
// Types for CDNA5 WMMA (wave32, V_WMMA_F32_16X16X32_BF16)
// ---------------------------------------------------------------------------
typedef __attribute__((ext_vector_type(16))) __bf16 v16bf;
typedef __attribute__((ext_vector_type(8)))  float  v8f;
typedef __attribute__((ext_vector_type(4)))  unsigned int u32x4;
typedef __attribute__((ext_vector_type(8)))  int i32x8;
typedef __attribute__((ext_vector_type(4)))  int i32x4;

union Frag {
  v16bf v;
  uint4 u[2];
};

__device__ __forceinline__ unsigned short f2bf(float f) {
  unsigned int u = __float_as_uint(f);
  u += 0x7FFFu + ((u >> 16) & 1u);   // round-to-nearest-even
  return (unsigned short)(u >> 16);
}

__device__ __forceinline__ v8f vzero8() { v8f z = {}; return z; }

// ---------------------------------------------------------------------------
// Tensor Data Mover: DMA a 2D tile of bf16 (2-byte elements) from global
// memory into LDS. Builds the D# descriptor per CDNA5 ISA ch.8:
//   group0: count=1, lds_addr, global_addr(57b), type=2
//   group1: data_size=1(2B), tensor_dim0/1, tile_dim0/1, tensor_dim0_stride
//   group2/3: zero (2D tensor)
// Completion tracked with TENSORcnt (s_wait_tensorcnt).
// ---------------------------------------------------------------------------
__device__ __forceinline__ void tdm_load2d_bf16(
    unsigned lds_addr, const void* gptr,
    unsigned tensor_d0, unsigned tensor_d1,
    unsigned tile_d0, unsigned tile_d1, unsigned stride0) {
  unsigned long long ga = (unsigned long long)(size_t)gptr;
  u32x4 g0;
  g0[0] = 1u;                                             // count=1, no gather
  g0[1] = lds_addr;                                       // LDS byte address
  g0[2] = (unsigned)(ga & 0xFFFFFFFFu);                   // global_addr[31:0]
  g0[3] = (unsigned)((ga >> 32) & 0x1FFFFFFu) | (2u << 30); // addr[56:32]|type=2
  i32x8 g1;
  g1[0] = (int)(1u << 16);                                // data_size=1 (2 bytes)
  g1[1] = (int)((tensor_d0 & 0xFFFFu) << 16);             // tensor_dim0[15:0]
  g1[2] = (int)((tensor_d0 >> 16) | ((tensor_d1 & 0xFFFFu) << 16));
  g1[3] = (int)((tensor_d1 >> 16) | (tile_d0 << 16));     // tile_dim0
  g1[4] = (int)(tile_d1 & 0xFFFFu);                       // tile_dim1, tile_dim2=0
  g1[5] = (int)stride0;                                   // tensor_dim0_stride lo
  g1[6] = 0;                                              // stride hi / dim1 stride
  g1[7] = 0;
  i32x4 gz = {0, 0, 0, 0};
#if __clang_major__ >= 23
  i32x8 gz8 = {0, 0, 0, 0, 0, 0, 0, 0};
  __builtin_amdgcn_tensor_load_to_lds(g0, g1, gz, gz, gz8, 0);
#else
  __builtin_amdgcn_tensor_load_to_lds(g0, g1, gz, gz, 0);
#endif
}

// ---------------------------------------------------------------------------
// fp32 -> bf16 elementwise convert
// ---------------------------------------------------------------------------
__global__ __launch_bounds__(256) void cvt_bf16_kernel(
    const float* __restrict__ in, unsigned short* __restrict__ out, size_t n) {
  size_t i = (size_t)blockIdx.x * 256 + threadIdx.x;
  if (i < n) out[i] = f2bf(in[i]);
}

// ---------------------------------------------------------------------------
// V: [B, T, KVH, 128] fp32  ->  [B, KVH, T, 128] bf16
// ---------------------------------------------------------------------------
__global__ __launch_bounds__(256) void v_relayout_kernel(
    const float* __restrict__ Vf, unsigned short* __restrict__ Vo, int T, size_t n) {
  size_t idx = (size_t)blockIdx.x * 256 + threadIdx.x;
  if (idx >= n) return;
  int d = (int)(idx & 127);
  size_t r = idx >> 7;
  int kvh = (int)(r & 3);
  size_t r2 = r >> 2;
  int t = (int)(r2 % T);
  int b = (int)(r2 / T);
  Vo[(((size_t)b * 4 + kvh) * T + t) * 128 + d] = f2bf(Vf[idx]);
}

// ---------------------------------------------------------------------------
// Fused RMS-norm + RoPE + gain + scale.
//   X: [B, T, NH, 128] fp32 (GEMM output layout)
//   O: [B, NH, T, 128] bf16 (attention layout)
// One wave (32 lanes) per head-row; lane owns rope pairs (l, l+64), (l+32, l+96).
// ---------------------------------------------------------------------------
__global__ __launch_bounds__(128) void rmsnorm_rope_kernel(
    const float* __restrict__ X, const float* __restrict__ gain,
    unsigned short* __restrict__ O, int T, int NH, float scale) {
  const int lane  = threadIdx.x & 31;
  const int rowid = blockIdx.x * 4 + (threadIdx.x >> 5);
  const int t = rowid % T;
  const int h = (rowid / T) % NH;
  const int b = rowid / (T * NH);
  const float* xp = X + (((size_t)b * T + t) * NH + h) * 128;

  const int i0 = lane, i1 = lane + 32;
  float x1a = xp[i0], x2a = xp[i0 + 64];
  float x1b = xp[i1], x2b = xp[i1 + 64];
  float ss = x1a * x1a + x2a * x2a + x1b * x1b + x2b * x2b;
#pragma unroll
  for (int m = 1; m <= 16; m <<= 1) ss += __shfl_xor(ss, m, 32);
  float rinv = rsqrtf(ss * (1.0f / 128.0f) + 1.1920929e-7f);
  float g = scale * (gain ? gain[h] : 1.0f) * rinv;

  const float lf = 0.14391156509897f;  // ln(10000)/64
  float ang_a = (float)t * __expf(-lf * (float)i0);
  float ang_b = (float)t * __expf(-lf * (float)i1);
  float sa, ca, sb, cb;
  sincosf(ang_a, &sa, &ca);
  sincosf(ang_b, &sb, &cb);

  unsigned short* op = O + (((size_t)b * NH + h) * T + t) * 128;
  op[i0]      = f2bf((x1a * ca + x2a * sa) * g);
  op[i0 + 64] = f2bf((-x1a * sa + x2a * ca) * g);
  op[i1]      = f2bf((x1b * cb + x2b * sb) * g);
  op[i1 + 64] = f2bf((-x1b * sb + x2b * cb) * g);
}

// ---------------------------------------------------------------------------
// GEMM: C[M,N] = A[M,K] * B[N,K]^T   (A,B bf16 row-major, C fp32)
// 256 threads = 8 waves; 128x128 tile; each wave -> 32x64 (2x4 WMMA tiles).
// A/B tiles staged in LDS by the Tensor Data Mover (double-buffered):
// wave 0 issues tensor_load_to_lds for tile i+1, waits TENSORcnt<=2 so tile i
// is resident, barrier, all waves run WMMA on tile i while tile i+1 streams.
// Fragment layouts per CDNA5 ISA 7.12.2:
//   A-frag (16x32): lane row = lane%16, kb = (lane/16)*8, k = {kb..kb+7, 16+kb..16+kb+7}
//   B-frag (32x16): lane col = lane%16, kb = (lane/16)*8, k = {kb..kb+7, 16+kb..16+kb+7}
// ---------------------------------------------------------------------------
__global__ __launch_bounds__(256) void gemm_bf16_nt(
    const unsigned short* __restrict__ A, const unsigned short* __restrict__ B,
    float* __restrict__ C, int M, int N, int K) {
  __shared__ __align__(16) unsigned short As[2][128][32];
  __shared__ __align__(16) unsigned short Bs[2][128][32];

  const int tid  = threadIdx.x;
  const int lane = tid & 31;
  const int wave = tid >> 5;
  const int wu   = __builtin_amdgcn_readfirstlane(wave);  // scalar wave id
  const int wm = wave & 3;   // 32-row block within tile
  const int wn = wave >> 2;  // 64-col block within tile
  const int m0 = blockIdx.y * 128;
  const int n0 = blockIdx.x * 128;
  const int lm = lane & 15;
  const int rs = lane >> 4;

  v8f acc[2][4];
#pragma unroll
  for (int i = 0; i < 2; ++i)
#pragma unroll
    for (int j = 0; j < 4; ++j) acc[i][j] = vzero8();

  const unsigned ldsA = (unsigned)(size_t)(void*)&As[0][0][0];
  const unsigned ldsB = (unsigned)(size_t)(void*)&Bs[0][0][0];
  const unsigned tileBytes = 128u * 32u * 2u;
  const int steps = K >> 5;

  if (wu == 0) {
    tdm_load2d_bf16(ldsA, A + (size_t)m0 * K, K, M, 32u, 128u, K);
    tdm_load2d_bf16(ldsB, B + (size_t)n0 * K, K, N, 32u, 128u, K);
  }

  for (int it = 0; it < steps; ++it) {
    const int cur = it & 1;
    if (wu == 0) {
      if (it + 1 < steps) {
        const int k1 = (it + 1) << 5;
        tdm_load2d_bf16(ldsA + (cur ^ 1) * tileBytes, A + (size_t)m0 * K + k1,
                        K, M, 32u, 128u, K);
        tdm_load2d_bf16(ldsB + (cur ^ 1) * tileBytes, B + (size_t)n0 * K + k1,
                        K, N, 32u, 128u, K);
        __builtin_amdgcn_s_wait_tensorcnt(2);  // tile `cur` (first pair) done
      } else {
        __builtin_amdgcn_s_wait_tensorcnt(0);
      }
    }
    __syncthreads();

    Frag af[2], bfr[4];
#pragma unroll
    for (int mt = 0; mt < 2; ++mt) {
      const unsigned short* p = &As[cur][wm * 32 + mt * 16 + lm][rs * 8];
      af[mt].u[0] = *(const uint4*)p;
      af[mt].u[1] = *(const uint4*)(p + 16);
    }
#pragma unroll
    for (int nt = 0; nt < 4; ++nt) {
      const unsigned short* p = &Bs[cur][wn * 64 + nt * 16 + lm][rs * 8];
      bfr[nt].u[0] = *(const uint4*)p;
      bfr[nt].u[1] = *(const uint4*)(p + 16);
    }
#pragma unroll
    for (int mt = 0; mt < 2; ++mt)
#pragma unroll
      for (int nt = 0; nt < 4; ++nt)
        acc[mt][nt] = __builtin_amdgcn_wmma_f32_16x16x32_bf16(
            false, af[mt].v, false, bfr[nt].v, (short)0, acc[mt][nt], false, false);
    __syncthreads();
  }

#pragma unroll
  for (int mt = 0; mt < 2; ++mt)
#pragma unroll
    for (int nt = 0; nt < 4; ++nt)
#pragma unroll
      for (int r = 0; r < 8; ++r) {
        int m = m0 + wm * 32 + mt * 16 + r + 8 * rs;
        int n = n0 + wn * 64 + nt * 16 + lm;
        C[(size_t)m * N + n] = acc[mt][nt][r];
      }
}

// ---------------------------------------------------------------------------
// Flash attention (causal, GQA group=4). Softmax scale folded into Q.
//   Q: [B, 16, T, 128] bf16, K/V: [B, 4, T, 128] bf16, Y: [B, T, 16*128] bf16
// Block = 128 threads (4 waves); block handles 64 q rows of one (b,h).
// Wave owns 16 q rows. K tile DMA'd to LDS by TDM (wave 0) while all
// waves stage V transposed; KV streamed in 64-key tiles.
// ---------------------------------------------------------------------------
__global__ __launch_bounds__(128) void flash_attn_kernel(
    const unsigned short* __restrict__ Q, const unsigned short* __restrict__ Kt,
    const unsigned short* __restrict__ Vt, unsigned short* __restrict__ Y, int T) {
  __shared__ __align__(16) unsigned short Ks[64][128];   // [key][d]
  __shared__ __align__(16) unsigned short Vst[128][64];  // [d][key] (transposed)
  __shared__ __align__(16) unsigned short Ps[4][16][64]; // per-wave P tile

  const int tid  = threadIdx.x;
  const int lane = tid & 31;
  const int wave = tid >> 5;
  const int wu   = __builtin_amdgcn_readfirstlane(wave);
  const int lm = lane & 15;
  const int rs = lane >> 4;
  const int qb = blockIdx.x;
  const int h  = blockIdx.y;
  const int b  = blockIdx.z;
  const int kvh = h >> 2;

  const size_t qbase = (((size_t)b * 16 + h) * T + (size_t)qb * 64 + wave * 16) * 128;
  const size_t kvbase = ((size_t)b * 4 + kvh) * (size_t)T * 128;
  const unsigned ldsK = (unsigned)(size_t)(void*)&Ks[0][0];

  // Preload Q A-fragments (4 k-steps over head_dim=128)
  Frag qf[4];
#pragma unroll
  for (int ks = 0; ks < 4; ++ks) {
    const unsigned short* p = Q + qbase + (size_t)lm * 128 + ks * 32 + rs * 8;
    qf[ks].u[0] = *(const uint4*)p;
    qf[ks].u[1] = *(const uint4*)(p + 16);
  }

  v8f oacc[8];
#pragma unroll
  for (int i = 0; i < 8; ++i) oacc[i] = vzero8();
  float m_i[8], l_i[8];
#pragma unroll
  for (int r = 0; r < 8; ++r) { m_i[r] = -3.0e38f; l_i[r] = 0.0f; }

  const int qrow0 = qb * 64 + wave * 16;

  for (int j = 0; j <= qb; ++j) {
    // --- K tile via Tensor Data Mover: 64 keys x 128 d, row-major ---
    if (wu == 0)
      tdm_load2d_bf16(ldsK, Kt + kvbase + (size_t)(j * 64) * 128,
                      128u, (unsigned)T, 128u, 64u, 128u);

    // --- V tile transposed to [d][key] by all threads ---
    const int krow = tid >> 1;
    const int dh = (tid & 1) * 64;
    const unsigned short* vp = Vt + kvbase + (size_t)(j * 64 + krow) * 128 + dh;
#pragma unroll
    for (int c = 0; c < 8; ++c) {
      uint4 tv = ((const uint4*)vp)[c];
      const unsigned short* ts = (const unsigned short*)&tv;
#pragma unroll
      for (int i = 0; i < 8; ++i) Vst[dh + c * 8 + i][krow] = ts[i];
    }
    if (wu == 0) __builtin_amdgcn_s_wait_tensorcnt(0);
    __syncthreads();

    // --- S = Q K^T (16 q-rows x 64 keys, per wave) ---
    v8f sacc[4];
#pragma unroll
    for (int t4 = 0; t4 < 4; ++t4) sacc[t4] = vzero8();
#pragma unroll
    for (int ks = 0; ks < 4; ++ks)
#pragma unroll
      for (int t4 = 0; t4 < 4; ++t4) {
        Frag bk;
        const unsigned short* p = &Ks[t4 * 16 + lm][ks * 32 + rs * 8];
        bk.u[0] = *(const uint4*)p;
        bk.u[1] = *(const uint4*)(p + 16);
        sacc[t4] = __builtin_amdgcn_wmma_f32_16x16x32_bf16(
            false, qf[ks].v, false, bk.v, (short)0, sacc[t4], false, false);
      }

    // --- causal mask on diagonal block only ---
    if (j == qb) {
#pragma unroll
      for (int t4 = 0; t4 < 4; ++t4)
#pragma unroll
        for (int r = 0; r < 8; ++r) {
          int mrow = qrow0 + r + 8 * rs;
          int key = j * 64 + t4 * 16 + lm;
          if (key > mrow) sacc[t4][r] = -3.0e38f;
        }
    }

    // --- online softmax (row lives across a 16-lane half in one VGPR slot) ---
    float alpha[8];
#pragma unroll
    for (int r = 0; r < 8; ++r) {
      float mx = sacc[0][r];
#pragma unroll
      for (int t4 = 1; t4 < 4; ++t4) mx = fmaxf(mx, sacc[t4][r]);
#pragma unroll
      for (int msk = 1; msk <= 8; msk <<= 1) mx = fmaxf(mx, __shfl_xor(mx, msk, 32));
      float mnew = fmaxf(m_i[r], mx);
      float al = __expf(m_i[r] - mnew);
      float rsum = 0.0f;
#pragma unroll
      for (int t4 = 0; t4 < 4; ++t4) {
        float p = __expf(sacc[t4][r] - mnew);
        sacc[t4][r] = p;
        rsum += p;
      }
#pragma unroll
      for (int msk = 1; msk <= 8; msk <<= 1) rsum += __shfl_xor(rsum, msk, 32);
      l_i[r] = l_i[r] * al + rsum;
      m_i[r] = mnew;
      alpha[r] = al;
    }
#pragma unroll
    for (int nt = 0; nt < 8; ++nt)
#pragma unroll
      for (int r = 0; r < 8; ++r) oacc[nt][r] *= alpha[r];

    // --- P: C-layout -> A-fragment layout via per-wave LDS ---
#pragma unroll
    for (int t4 = 0; t4 < 4; ++t4)
#pragma unroll
      for (int r = 0; r < 8; ++r)
        Ps[wave][r + 8 * rs][t4 * 16 + lm] = f2bf(sacc[t4][r]);
    __syncthreads();

    // --- O += P V  (2 k-steps over 64 keys; 8 d-tiles) ---
#pragma unroll
    for (int ks2 = 0; ks2 < 2; ++ks2) {
      Frag pa;
      const unsigned short* pp = &Ps[wave][lm][ks2 * 32 + rs * 8];
      pa.u[0] = *(const uint4*)pp;
      pa.u[1] = *(const uint4*)(pp + 16);
#pragma unroll
      for (int nt = 0; nt < 8; ++nt) {
        Frag vb;
        const unsigned short* vpp = &Vst[nt * 16 + lm][ks2 * 32 + rs * 8];
        vb.u[0] = *(const uint4*)vpp;
        vb.u[1] = *(const uint4*)(vpp + 16);
        oacc[nt] = __builtin_amdgcn_wmma_f32_16x16x32_bf16(
            false, pa.v, false, vb.v, (short)0, oacc[nt], false, false);
      }
    }
    __syncthreads();
  }

  // --- epilogue: normalize and write y in [B, T, H, 128] bf16 ---
#pragma unroll
  for (int r = 0; r < 8; ++r) {
    float inv = 1.0f / l_i[r];
    int trow = qrow0 + r + 8 * rs;
    unsigned short* yp = Y + (((size_t)b * T + trow) * 16 + h) * 128;
#pragma unroll
    for (int nt = 0; nt < 8; ++nt)
      yp[nt * 16 + lm] = f2bf(oacc[nt][r] * inv);
  }
}

// ---------------------------------------------------------------------------
// Host launcher
// ---------------------------------------------------------------------------
extern "C" void kernel_launch(void* const* d_in, const int* in_sizes, int n_in,
                              void* d_out, int out_size, void* d_ws, size_t ws_size,
                              hipStream_t stream) {
  (void)in_sizes; (void)n_in; (void)out_size; (void)ws_size;
  const int B = 2, T = 2048, D = 2048, H = 16, KVH = 4;
  const int M = B * T;        // 4096
  const int KVD = KVH * 128;  // 512
  const float attn_scale = 0.08838834764831845f;  // 1/sqrt(128)

  const float* x  = (const float*)d_in[0];
  const float* qg = (const float*)d_in[1];
  const float* Wq = (const float*)d_in[2];
  const float* Wk = (const float*)d_in[3];
  const float* Wv = (const float*)d_in[4];
  const float* Wp = (const float*)d_in[5];
  float* out = (float*)d_out;

  char* ws = (char*)d_ws;
  size_t off = 0;
  auto alloc = [&](size_t bytes) -> void* {
    void* p = ws + off;
    off = (off + bytes + 255) & ~(size_t)255;
    return p;
  };

  unsigned short* xb  = (unsigned short*)alloc((size_t)M * D * 2);
  unsigned short* wqb = (unsigned short*)alloc((size_t)D * D * 2);
  unsigned short* wkb = (unsigned short*)alloc((size_t)KVD * D * 2);
  unsigned short* wvb = (unsigned short*)alloc((size_t)KVD * D * 2);
  unsigned short* wpb = (unsigned short*)alloc((size_t)D * D * 2);
  float* qf = (float*)alloc((size_t)M * D * 4);
  float* kf = (float*)alloc((size_t)M * KVD * 4);
  float* vf = (float*)alloc((size_t)M * KVD * 4);
  unsigned short* qb16 = (unsigned short*)alloc((size_t)M * D * 2);
  unsigned short* kb16 = (unsigned short*)alloc((size_t)M * KVD * 2);
  unsigned short* vb16 = (unsigned short*)alloc((size_t)M * KVD * 2);
  unsigned short* yb16 = (unsigned short*)alloc((size_t)M * D * 2);

  auto cvt = [&](const float* src, unsigned short* dst, size_t n) {
    cvt_bf16_kernel<<<(unsigned)((n + 255) / 256), 256, 0, stream>>>(src, dst, n);
  };
  cvt(x,  xb,  (size_t)M * D);
  cvt(Wq, wqb, (size_t)D * D);
  cvt(Wk, wkb, (size_t)KVD * D);
  cvt(Wv, wvb, (size_t)KVD * D);
  cvt(Wp, wpb, (size_t)D * D);

  // Projections: q = x Wq^T, k = x Wk^T, v = x Wv^T
  gemm_bf16_nt<<<dim3(D / 128, M / 128), 256, 0, stream>>>(xb, wqb, qf, M, D, D);
  gemm_bf16_nt<<<dim3(KVD / 128, M / 128), 256, 0, stream>>>(xb, wkb, kf, M, KVD, D);
  gemm_bf16_nt<<<dim3(KVD / 128, M / 128), 256, 0, stream>>>(xb, wvb, vf, M, KVD, D);

  // RMS-norm + RoPE (+ gain + softmax scale for Q)
  rmsnorm_rope_kernel<<<(B * H * T) / 4, 128, 0, stream>>>(qf, qg, qb16, T, H, attn_scale);
  rmsnorm_rope_kernel<<<(B * KVH * T) / 4, 128, 0, stream>>>(kf, nullptr, kb16, T, KVH, 1.0f);
  v_relayout_kernel<<<(unsigned)(((size_t)M * KVD + 255) / 256), 256, 0, stream>>>(
      vf, vb16, T, (size_t)M * KVD);

  // Causal flash attention
  flash_attn_kernel<<<dim3(T / 64, H, B), 128, 0, stream>>>(qb16, kb16, vb16, yb16, T);

  // Output projection -> fp32 d_out
  gemm_bf16_nt<<<dim3(D / 128, M / 128), 256, 0, stream>>>(yb16, wpb, out, M, D, D);
}
